// MultiHeadAttentionWithEntropy_19679540150533
// MI455X (gfx1250) — matrix operand order repaired and checked by reading, not compiled, not run on hardware
//
#include <hip/hip_runtime.h>
#include <stdint.h>

// ---------- types ----------
typedef __attribute__((ext_vector_type(16))) __bf16 v16bf;
typedef __attribute__((ext_vector_type(8)))  float  v8f;
typedef __attribute__((ext_vector_type(4)))  unsigned int u32x4;
typedef __attribute__((ext_vector_type(4)))  float  f32x4;

#define DEV __device__ __forceinline__

static constexpr int Bb = 2, T = 2048, D = 1024, H = 16, HD = 64;
static constexpr int M_TOT = Bb * T;           // 4096
static constexpr float SCALE = 0.125f;         // 1/sqrt(64)

// ---------- helpers ----------
DEV __bf16 f2bf(float f) {
  union { float f; uint32_t u; } v; v.f = f;
  uint32_t r = v.u + 0x7FFFu + ((v.u >> 16) & 1u);   // round-to-nearest-even
  union { unsigned short s; __bf16 b; } o; o.s = (unsigned short)(r >> 16);
  return o.b;
}
DEV unsigned short f2bf_bits(float f) {
  union { __bf16 b; unsigned short s; } o; o.b = f2bf(f); return o.s;
}
DEV unsigned int pack2(float a, float b) {
  union { unsigned short s[2]; unsigned int u; } p;
  p.s[0] = f2bf_bits(a); p.s[1] = f2bf_bits(b); return p.u;
}

// low 32 bits of a generic shared-memory pointer == LDS byte offset
DEV unsigned lds_u32(const void* p) { return (unsigned)(size_t)p; }

// gfx1250 async DMA: global -> LDS, 16B per lane, tracked by ASYNCcnt
DEV void async_ld_b128(unsigned lds_off, const void* gptr) {
  asm volatile("global_load_async_to_lds_b128 %0, %1, off"
               :: "v"(lds_off), "v"(gptr)
               : "memory");
}
DEV void wait_async0() {
  asm volatile("s_wait_asynccnt 0x0" ::: "memory");
}

union FragAB { u32x4 u[2]; v16bf v; };

DEV v8f wmma_bf16(const FragAB& a, const FragAB& b, v8f c) {
  // D = A(16x32 bf16) x B(32x16 bf16) + C(16x16 f32)
  return __builtin_amdgcn_wmma_f32_16x16x32_bf16(
      false, a.v, false, b.v, (short)0, c, false, false);
}

// =====================================================================
// Elementwise f32 -> bf16 (8 elems / thread, b128 in / b128 out)
// =====================================================================
__global__ __launch_bounds__(256) void cvt_bf16_kernel(
    const float* __restrict__ in, unsigned short* __restrict__ out, int n8)
{
  int i = blockIdx.x * 256 + threadIdx.x;
  if (i < n8) {
    f32x4 a = *(const f32x4*)(in + (size_t)i * 8);
    f32x4 b = *(const f32x4*)(in + (size_t)i * 8 + 4);
    u32x4 o;
    o.x = pack2(a.x, a.y); o.y = pack2(a.z, a.w);
    o.z = pack2(b.x, b.y); o.w = pack2(b.z, b.w);
    *(u32x4*)(out + (size_t)i * 8) = o;
  }
}

// =====================================================================
// GEMM (all-bf16 inputs): Out[M,N] = A[M,K] @ W[N,K]^T
// Block tile 128x128, wave tile 32x64 (2x4 WMMA), K-step 32,
// double-buffered async-to-LDS staging (8 WMMA per 12 LDS b128 reads).
// SPLIT_HEADS: write bf16 to [B,H,T,64]; else f32 row-major [M,N].
// =====================================================================
template<bool SPLIT_HEADS>
__global__ __launch_bounds__(256) void gemm_bf16_nt_kernel(
    const unsigned short* __restrict__ A_, const unsigned short* __restrict__ W_,
    void* __restrict__ Out, int M, int N, int K)
{
  __shared__ __attribute__((aligned(16))) __bf16 As[2][128][40]; // 80B rows
  __shared__ __attribute__((aligned(16))) __bf16 Bs[2][128][40];

  const __bf16* A = (const __bf16*)A_;
  const __bf16* W = (const __bf16*)W_;

  const int tid  = threadIdx.x;
  const int wave = tid >> 5, lane = tid & 31;
  const int half = lane >> 4, ln = lane & 15;
  const int wm = (wave & 3) * 32;        // 4 waves along M
  const int wn = (wave >> 2) * 64;       // 2 waves along N
  const int bm = blockIdx.x * 128, bn = blockIdx.y * 128;
  const int NK = K / 32;

  v8f acc[2][4];
  #pragma unroll
  for (int i = 0; i < 2; i++)
    #pragma unroll
    for (int j = 0; j < 4; j++)
      #pragma unroll
      for (int e = 0; e < 8; e++) acc[i][j][e] = 0.f;

  auto stage = [&](int kk) {
    const int buf = kk & 1, k0 = kk * 32;
    #pragma unroll
    for (int i = 0; i < 2; i++) {                    // A & B: 512 x 16B each
      int slot = tid + i * 256;
      int row = slot >> 2, col = (slot & 3) << 3;
      async_ld_b128(lds_u32(&As[buf][row][col]),
                    A + (size_t)(bm + row) * K + k0 + col);
      async_ld_b128(lds_u32(&Bs[buf][row][col]),
                    W + (size_t)(bn + row) * K + k0 + col);
    }
  };

  stage(0);
  wait_async0();
  __syncthreads();

  for (int kk = 0; kk < NK; kk++) {
    if (kk + 1 < NK) stage(kk + 1);                  // prefetch next slab
    const int buf = kk & 1;

    FragAB af[2], bfr[4];
    #pragma unroll
    for (int mi = 0; mi < 2; mi++) {
      af[mi].u[0] = *(const u32x4*)&As[buf][wm + mi * 16 + ln][half * 8];
      af[mi].u[1] = *(const u32x4*)&As[buf][wm + mi * 16 + ln][16 + half * 8];
    }
    #pragma unroll
    for (int ni = 0; ni < 4; ni++) {
      bfr[ni].u[0] = *(const u32x4*)&Bs[buf][wn + ni * 16 + ln][half * 8];
      bfr[ni].u[1] = *(const u32x4*)&Bs[buf][wn + ni * 16 + ln][16 + half * 8];
    }
    #pragma unroll
    for (int mi = 0; mi < 2; mi++)
      #pragma unroll
      for (int ni = 0; ni < 4; ni++)
        acc[mi][ni] = wmma_bf16(af[mi], bfr[ni], acc[mi][ni]);

    wait_async0();          // next slab landed in LDS
    __syncthreads();        // visible to all waves; cur slab reads done
  }

  // ---- writeback (C layout: M = e + 8*half, N = lane&15) ----
  #pragma unroll
  for (int mi = 0; mi < 2; mi++)
    #pragma unroll
    for (int ni = 0; ni < 4; ni++)
      #pragma unroll
      for (int e = 0; e < 8; e++) {
        int gm = bm + wm + mi * 16 + e + half * 8;
        int gn = bn + wn + ni * 16 + ln;
        float val = acc[mi][ni][e];
        if (SPLIT_HEADS) {
          int b = gm >> 11, t = gm & (T - 1);
          int h = gn >> 6, d = gn & 63;
          ((unsigned short*)Out)[((((size_t)b * H + h) * T + t) << 6) + d] =
              f2bf_bits(val);
        } else {
          ((float*)Out)[(size_t)gm * N + gn] = val;
        }
      }
}

// =====================================================================
// Flash attention per (b, h, 64-row q tile), online softmax + entropy.
// 8 waves; wave w owns rows 16*(w>>1), cols 32*(w&1). K/V double-buffered,
// K via async-to-LDS, V transposed via VALU. Softmax: 4 threads / row.
// =====================================================================
__global__ __launch_bounds__(256) void attn_kernel(
    const unsigned short* __restrict__ Qh_, const unsigned short* __restrict__ Kh_,
    const unsigned short* __restrict__ Vh_, const unsigned char* __restrict__ mask,
    unsigned short* __restrict__ Om, float* __restrict__ entAcc)
{
  __shared__ __attribute__((aligned(16))) __bf16 Qs[64][72];
  __shared__ __attribute__((aligned(16))) __bf16 Ks[2][64][72];   // [key][d]
  __shared__ __attribute__((aligned(16))) __bf16 VsT[2][64][72];  // [d][key]
  __shared__ __attribute__((aligned(16))) __bf16 Ps[64][72];
  __shared__ float  Ss[64][65];
  __shared__ float  pmax[64][4], pZ[64][4], pS[64][4];
  __shared__ float  alphaS[64];
  __shared__ float  invZs[64];
  __shared__ float  entS[64];

  const int tid  = threadIdx.x;
  const int wave = tid >> 5, lane = tid & 31;
  const int half = lane >> 4, ln = lane & 15;
  const int rowb = (wave >> 1) * 16;
  const int colb = (wave & 1) * 32;
  const int r  = tid >> 2;            // softmax: row 0..63
  const int qd = tid & 3;             // softmax: quarter
  const int c0 = qd * 16;

  const int q0 = blockIdx.x * 64;
  const int h  = blockIdx.y;
  const int b  = blockIdx.z;

  const __bf16* Qbase = (const __bf16*)Qh_ + (((size_t)b * H + h) * T + q0) * HD;
  const __bf16* Kbase = (const __bf16*)Kh_ + (((size_t)b * H + h) * T) * HD;
  const __bf16* Vbase = (const __bf16*)Vh_ + (((size_t)b * H + h) * T) * HD;
  const unsigned char* mbase = mask + ((size_t)b * T + q0 + r) * T + c0;

  auto stageKV = [&](int j) {
    const int buf = j & 1;
    #pragma unroll
    for (int i = 0; i < 2; i++) {                    // K: async DMA copy
      int slot = tid + i * 256;
      int key = slot >> 3, dcol = (slot & 7) << 3;
      async_ld_b128(lds_u32(&Ks[buf][key][dcol]),
                    Kbase + ((size_t)(j * 64 + key)) * HD + dcol);
    }
    // V transposed: thread covers keys {2kp, 2kp+1} x 8 d-cols -> b32 stores
    int kp = tid >> 3, g = (tid & 7) << 3;
    u32x4 v0 = *(const u32x4*)(Vbase + ((size_t)(j * 64 + 2 * kp)) * HD + g);
    u32x4 v1 = *(const u32x4*)(Vbase + ((size_t)(j * 64 + 2 * kp + 1)) * HD + g);
    union { u32x4 q; unsigned short hh[8]; } a, c; a.q = v0; c.q = v1;
    #pragma unroll
    for (int u = 0; u < 8; u++) {
      unsigned int pk = (unsigned)a.hh[u] | ((unsigned)c.hh[u] << 16);
      *(unsigned int*)&VsT[buf][g + u][2 * kp] = pk;
    }
  };

  // stage Q (async) + first K/V tile
  #pragma unroll
  for (int i = 0; i < 2; i++) {
    int slot = tid + i * 256;
    int row = slot >> 3, col = (slot & 7) << 3;
    async_ld_b128(lds_u32(&Qs[row][col]), Qbase + (size_t)row * HD + col);
  }
  stageKV(0);
  wait_async0();
  __syncthreads();

  v8f oacc[2];
  #pragma unroll
  for (int ni = 0; ni < 2; ni++)
    #pragma unroll
    for (int e = 0; e < 8; e++) oacc[ni][e] = 0.f;

  // per-row online state, replicated in the 4 owner threads of each row
  float m_r = -3.0e38f, Z_r = 0.f, SS_r = 0.f;

  const int NT = T / 64;
  for (int j = 0; j < NT; j++) {
    if (j + 1 < NT) stageKV(j + 1);                // prefetch next tile
    const int buf = j & 1;

    // ---- S = Q @ K^T ----
    v8f sacc[2];
    #pragma unroll
    for (int ni = 0; ni < 2; ni++)
      #pragma unroll
      for (int e = 0; e < 8; e++) sacc[ni][e] = 0.f;

    #pragma unroll
    for (int d0 = 0; d0 < 64; d0 += 32) {
      FragAB af, bfr;
      af.u[0] = *(const u32x4*)&Qs[rowb + ln][d0 + half * 8];
      af.u[1] = *(const u32x4*)&Qs[rowb + ln][d0 + 16 + half * 8];
      #pragma unroll
      for (int ni = 0; ni < 2; ni++) {
        bfr.u[0] = *(const u32x4*)&Ks[buf][colb + ni * 16 + ln][d0 + half * 8];
        bfr.u[1] = *(const u32x4*)&Ks[buf][colb + ni * 16 + ln][d0 + 16 + half * 8];
        sacc[ni] = wmma_bf16(af, bfr, sacc[ni]);
      }
    }
    #pragma unroll
    for (int ni = 0; ni < 2; ni++)
      #pragma unroll
      for (int e = 0; e < 8; e++)
        Ss[rowb + e + half * 8][colb + ni * 16 + ln] = sacc[ni][e];
    __syncthreads();

    // ---- online softmax + entropy, 4 threads per row ----
    {
      const unsigned char* mrow = mbase + (size_t)j * 64;
      float lmax = -3.0e38f;
      #pragma unroll
      for (int c = 0; c < 16; c++) {
        float s = Ss[r][c0 + c] * SCALE;
        if (!mrow[c]) s = -1.0e9f;
        Ss[r][c0 + c] = s;
        lmax = fmaxf(lmax, s);
      }
      pmax[r][qd] = lmax;
    }
    __syncthreads();
    float m_save, alpha_save;
    {
      float mt = fmaxf(fmaxf(pmax[r][0], pmax[r][1]),
                       fmaxf(pmax[r][2], pmax[r][3]));
      m_save = fmaxf(m_r, mt);
      alpha_save = __expf(m_r - m_save);
      float Zt = 0.f, St = 0.f;
      #pragma unroll
      for (int c = 0; c < 16; c++) {
        float s = Ss[r][c0 + c];
        float e = __expf(s - m_save);
        Zt += e; St += e * s;
        Ps[r][c0 + c] = f2bf(e);
      }
      pZ[r][qd] = Zt; pS[r][qd] = St;
      if (qd == 0) alphaS[r] = alpha_save;
    }
    __syncthreads();
    {
      float Zt = (pZ[r][0] + pZ[r][1]) + (pZ[r][2] + pZ[r][3]);
      float St = (pS[r][0] + pS[r][1]) + (pS[r][2] + pS[r][3]);
      Z_r  = Z_r * alpha_save + Zt;
      SS_r = SS_r * alpha_save + St;
      m_r  = m_save;
    }

    // ---- rescale O, then O += P @ V ----
    #pragma unroll
    for (int ni = 0; ni < 2; ni++)
      #pragma unroll
      for (int e = 0; e < 8; e++)
        oacc[ni][e] *= alphaS[rowb + e + half * 8];

    #pragma unroll
    for (int k0 = 0; k0 < 64; k0 += 32) {
      FragAB af, bfr;
      af.u[0] = *(const u32x4*)&Ps[rowb + ln][k0 + half * 8];
      af.u[1] = *(const u32x4*)&Ps[rowb + ln][k0 + 16 + half * 8];
      #pragma unroll
      for (int ni = 0; ni < 2; ni++) {
        bfr.u[0] = *(const u32x4*)&VsT[buf][colb + ni * 16 + ln][k0 + half * 8];
        bfr.u[1] = *(const u32x4*)&VsT[buf][colb + ni * 16 + ln][k0 + 16 + half * 8];
        oacc[ni] = wmma_bf16(af, bfr, oacc[ni]);
      }
    }

    wait_async0();          // next K tile landed
    __syncthreads();        // visible; cur buffers free for reuse
  }

  if (qd == 0) {
    float iz = 1.0f / Z_r;
    invZs[r] = iz;
    entS[r]  = m_r + __logf(Z_r) - SS_r * iz;
  }
  __syncthreads();
  if (tid == 0) {
    float s = 0.f;
    for (int i = 0; i < 64; i++) s += entS[i];
    atomicAdd(entAcc, s);
  }

  // ---- write O tile, merged heads: Om[b][t][h*64+c] (bf16) ----
  const size_t orow = (size_t)b * T + q0;
  #pragma unroll
  for (int ni = 0; ni < 2; ni++)
    #pragma unroll
    for (int e = 0; e < 8; e++) {
      int rr = rowb + e + half * 8;
      int cc = colb + ni * 16 + ln;
      float val = oacc[ni][e] * invZs[rr];
      Om[(orow + rr) * D + h * HD + cc] = f2bf_bits(val);
    }
}

// =====================================================================
__global__ void zero_kernel(float* p) { p[0] = 0.f; }

__global__ void finalize_kernel(const float* entAcc, float* y) {
  y[(size_t)M_TOT * D] = entAcc[0] * (1.0f / 65536.0f);  // mean over B*H*T
}

// =====================================================================
extern "C" void kernel_launch(void* const* d_in, const int* in_sizes, int n_in,
                              void* d_out, int out_size, void* d_ws, size_t ws_size,
                              hipStream_t stream) {
  const float* q  = (const float*)d_in[0];
  const float* k  = (const float*)d_in[1];
  const float* v  = (const float*)d_in[2];
  const unsigned char* mask = (const unsigned char*)d_in[3];
  const float* Wq = (const float*)d_in[4];
  const float* Wk = (const float*)d_in[5];
  const float* Wv = (const float*)d_in[6];
  const float* Wc = (const float*)d_in[7];
  float* y = (float*)d_out;

  const size_t actElems = (size_t)M_TOT * D;          // 4,194,304
  const size_t wElems   = (size_t)D * D;              // 1,048,576
  char* ws = (char*)d_ws;
  float* entAcc       = (float*)ws;
  unsigned short* Qh  = (unsigned short*)(ws + 256);
  unsigned short* Kh  = Qh + actElems;
  unsigned short* Vh  = Kh + actElems;
  unsigned short* Om  = Vh + actElems;
  unsigned short* qb  = Om + actElems;
  unsigned short* kb  = qb + actElems;
  unsigned short* vb  = kb + actElems;
  unsigned short* Wqb = vb + actElems;
  unsigned short* Wkb = Wqb + wElems;
  unsigned short* Wvb = Wkb + wElems;
  unsigned short* Wcb = Wvb + wElems;

  dim3 blk(256);
  dim3 gProj(M_TOT / 128, D / 128);                   // 32 x 8

  zero_kernel<<<1, 1, 0, stream>>>(entAcc);
  // f32 -> bf16 copies (activations + weights)
  cvt_bf16_kernel<<<(int)(actElems / 8 / 256), blk, 0, stream>>>(q, qb, (int)(actElems / 8));
  cvt_bf16_kernel<<<(int)(actElems / 8 / 256), blk, 0, stream>>>(k, kb, (int)(actElems / 8));
  cvt_bf16_kernel<<<(int)(actElems / 8 / 256), blk, 0, stream>>>(v, vb, (int)(actElems / 8));
  cvt_bf16_kernel<<<(int)(wElems / 8 / 256), blk, 0, stream>>>(Wq, Wqb, (int)(wElems / 8));
  cvt_bf16_kernel<<<(int)(wElems / 8 / 256), blk, 0, stream>>>(Wk, Wkb, (int)(wElems / 8));
  cvt_bf16_kernel<<<(int)(wElems / 8 / 256), blk, 0, stream>>>(Wv, Wvb, (int)(wElems / 8));
  cvt_bf16_kernel<<<(int)(wElems / 8 / 256), blk, 0, stream>>>(Wc, Wcb, (int)(wElems / 8));
  // projections (bf16 WMMA, head-split bf16 outputs)
  gemm_bf16_nt_kernel<true><<<gProj, blk, 0, stream>>>(qb, Wqb, Qh, M_TOT, D, D);
  gemm_bf16_nt_kernel<true><<<gProj, blk, 0, stream>>>(kb, Wkb, Kh, M_TOT, D, D);
  gemm_bf16_nt_kernel<true><<<gProj, blk, 0, stream>>>(vb, Wvb, Vh, M_TOT, D, D);
  // flash attention + entropy
  attn_kernel<<<dim3(T / 64, H, Bb), blk, 0, stream>>>(Qh, Kh, Vh, mask, Om, entAcc);
  // output projection (f32 out)
  gemm_bf16_nt_kernel<false><<<gProj, blk, 0, stream>>>(Om, Wcb, y, M_TOT, D, D);
  finalize_kernel<<<1, 1, 0, stream>>>(entAcc, y);
}
